// MiniCompletionNet_43095701848490
// MI455X (gfx1250) — matrix operand order repaired and checked by reading, not compiled, not run on hardware
//
#include <hip/hip_runtime.h>
#include <hip/hip_bf16.h>

// ---------------------------------------------------------------------------
// MiniCompletionNet on MI455X (gfx1250).
// Compute-bound (~400 GFLOP vs ~20us of HBM traffic at 23.3 TB/s), so all
// conv layers run as implicit GEMM on V_WMMA_F32_16X16X32_F16 (f16 inputs,
// f32 accumulate; per-channel BN right after each conv absorbs f16 input
// quantization). BN/ELU/skips in f32, deterministic tree reductions.
// Weight tiles are staged LDS-side with GLOBAL_LOAD_ASYNC_TO_LDS_B128
// (ASYNCcnt + s_wait_asynccnt); im2col gather is branch-free selects.
// ---------------------------------------------------------------------------

typedef _Float16 h16;
typedef __attribute__((ext_vector_type(16))) _Float16 v16h;
typedef __attribute__((ext_vector_type(8)))  _Float16 v8h;
typedef __attribute__((ext_vector_type(8)))  float    v8f;

#define A_STRIDE 40   // halves per LDS row: 32 data + 8 pad (80B, conflict-free)
#define B_STRIDE 40

// ---------------------------------------------------------------------------
// Implicit-GEMM 3D convolution.
//   X : (Nb, Ctot, Din,Hin,Win) f16, channels [coff, coff+Cin) used
//   Wt: (Cout, Cin*k^3) f16 (OIDHW flattened)
//   Y : (Nb, Cout, Do,Ho,Wo) f32
// Block: 128 threads = 4 wave32. Block tile: 64 (Cout) x 64 (spatial).
// Each wave owns a 16-row M strip and all 64 N columns (4 WMMA tiles).
// aok32: K % 32 == 0 -> A tile staged via async global->LDS b128 copies.
// Out-of-range Cout rows read a clamped (row 0) pointer; their products are
// finite garbage and are never stored, so staging needs no per-lane predicate.
// ---------------------------------------------------------------------------
__global__ __launch_bounds__(128) void conv3d_wmma_kernel(
    const h16* __restrict__ X, const h16* __restrict__ Wt, float* __restrict__ Y,
    int Nb, int Cin, int Ctot, int coff,
    int Din, int Hin, int Win,
    int Cout, int K, int k, int k2, int k3,
    int stride, int padlo,
    int Do, int Ho, int Wo, int aok32)
{
  __shared__ __attribute__((aligned(16))) h16 As[64 * A_STRIDE];
  __shared__ __attribute__((aligned(16))) h16 Bs[64 * B_STRIDE];

  const int S   = Do * Ho * Wo;
  const int Ns  = Nb * S;
  const int Sin = Din * Hin * Win;
  const int tid  = threadIdx.x;
  const int lane = tid & 31;
  const int wave = tid >> 5;
  const int m0 = blockIdx.y * 64;
  const int n0 = blockIdx.x * 64;

  const v8f vzero = {0.f,0.f,0.f,0.f,0.f,0.f,0.f,0.f};
  v8f acc[4] = {vzero, vzero, vzero, vzero};

  // --- per-thread B-gather coordinates: this thread fills Bs[nn][kk0..kk0+15]
  const int nn  = tid >> 1;
  const int kk0 = (tid & 1) * 16;
  const int sg  = n0 + nn;            // global spatial-GEMM column
  int bn = 0, bz = 0, by = 0, bx = 0;
  const bool svalid = sg < Ns;
  if (svalid) {
    bn = sg / S;            int r  = sg - bn * S;
    bz = r / (Ho * Wo);     int r2 = r - bz * (Ho * Wo);
    by = r2 / Wo;           bx = r2 - by * Wo;
  }
  const int ozb = bz * stride - padlo;
  const int oyb = by * stride - padlo;
  const int oxb = bx * stride - padlo;
  const h16* xbase = X + (size_t)bn * Ctot * Sin + (size_t)coff * Sin;

  // --- per-thread A coordinates: this thread fills As[arow][ak0..ak0+15]
  const int arow = tid >> 1;
  const int ak0  = (tid & 1) * 16;
  const bool avalid = (m0 + arow) < Cout;
  // clamped row pointer: always dereferenceable; invalid rows never stored
  const h16* wrow = Wt + (size_t)(avalid ? (m0 + arow) : 0) * K;
  // LDS byte offset of this thread's A-tile slot (low 32 bits of generic ptr)
  const unsigned ldsA = (unsigned)(size_t)(void*)&As[arow * A_STRIDE + ak0];

  for (int kc = 0; kc < K; kc += 32) {
    // -------- stage A (weights) tile: 64 x 32 halves
    if (aok32) {
      // uniform branch, EXEC all ones: async DMA 32B/thread global->LDS.
      // INST_OFFSET is added to BOTH the LDS and global addresses, so one
      // VDST register covers both 16B halves.
      const h16* gp = wrow + kc + ak0;
      __builtin_prefetch(gp + 64, 0, 1);                   // global_prefetch_b8
      asm volatile(
          "global_load_async_to_lds_b128 %0, %1, off\n\t"
          "global_load_async_to_lds_b128 %0, %1, off offset:16"
          :
          : "v"(ldsA), "v"(gp)
          : "memory");
    } else {
      v8h a0, a1;
      #pragma unroll
      for (int j = 0; j < 16; ++j) {
        int kg  = kc + ak0 + j;
        bool ok = kg < K;
        int kgc = ok ? kg : 0;                             // clamp, no branch
        h16 raw = wrow[kgc];
        h16 v   = ok ? raw : (h16)0;
        if (j < 8) a0[j] = v; else a1[j - 8] = v;
      }
      *(v8h*)&As[arow * A_STRIDE + ak0]     = a0;
      *(v8h*)&As[arow * A_STRIDE + ak0 + 8] = a1;
    }

    // -------- stage B (im2col) tile: 32(K) x 64(N), stored [col][kk]
    // one decompose per chunk, then a branch-free carry chain per element
    {
      int kg = kc + kk0;
      int ci = kg / k3;  int r  = kg - ci * k3;
      int kz = r  / k2;  int r2 = r  - kz * k2;
      int ky = r2 / k;   int kx = r2 - ky * k;
      v8h b0, b1;
      #pragma unroll
      for (int j = 0; j < 16; ++j) {
        int iz = ozb + kz, iy = oyb + ky, ix = oxb + kx;
        bool ok = svalid && (kg + j < K) &&
                  (unsigned)iz < (unsigned)Din &&
                  (unsigned)iy < (unsigned)Hin &&
                  (unsigned)ix < (unsigned)Win;
        size_t offv = ok ? ((size_t)ci * Sin + (iz * Hin + iy) * Win + ix)
                         : (size_t)0;                      // clamp, no branch
        h16 raw = xbase[offv];
        h16 v   = ok ? raw : (h16)0;
        if (j < 8) b0[j] = v; else b1[j - 8] = v;
        // advance (ci,kz,ky,kx) by one K step: select-only carry chain
        kx += 1;
        int c1 = (kx == k) ? 1 : 0; kx = c1 ? 0 : kx; ky += c1;
        int c2 = (ky == k) ? 1 : 0; ky = c2 ? 0 : ky; kz += c2;
        int c3 = (kz == k) ? 1 : 0; kz = c3 ? 0 : kz; ci += c3;
      }
      *(v8h*)&Bs[nn * B_STRIDE + kk0]     = b0;            // ds_store_b128
      *(v8h*)&Bs[nn * B_STRIDE + kk0 + 8] = b1;
    }

    // drain async A-tile copies, then block barrier
    if (aok32) asm volatile("s_wait_asynccnt 0" ::: "memory");
    __syncthreads();

    // -------- WMMA: fragments per ISA 16-bit layouts
    // A 16x32: lanes 0-15 hold K kb..kb+7 (v0-3) and kb+16..kb+23 (v4-7), kb=(lane>>4)*8
    // B 32x16: lane's column = lane&15, K kb..kb+7 and kb+16..kb+23, kb=(lane>>4)*8
    const int half = lane >> 4;
    const int l15  = lane & 15;
    const int arw  = wave * 16 + l15;
    const int kbA  = half * 8;
    const int kbB  = half * 8;

    v8h alo = *(const v8h*)&As[arw * A_STRIDE + kbA];
    v8h ahi = *(const v8h*)&As[arw * A_STRIDE + kbA + 16];
    v16h a;
    #pragma unroll
    for (int i = 0; i < 8; ++i) { a[i] = alo[i]; a[i + 8] = ahi[i]; }

    #pragma unroll
    for (int nt = 0; nt < 4; ++nt) {
      int col = nt * 16 + l15;
      v8h blo = *(const v8h*)&Bs[col * B_STRIDE + kbB];
      v8h bhi = *(const v8h*)&Bs[col * B_STRIDE + kbB + 16];
      v16h b;
      #pragma unroll
      for (int i = 0; i < 8; ++i) { b[i] = blo[i]; b[i + 8] = bhi[i]; }
      acc[nt] = __builtin_amdgcn_wmma_f32_16x16x32_f16(
          false, a, false, b, (short)0, acc[nt], false, false);
    }
    __syncthreads();
  }

  // -------- store D: f32 16x16 layout: VGPR r -> M = r + (lane>>4)*8, N = lane&15
  const int half = lane >> 4;
  const int l15  = lane & 15;
  #pragma unroll
  for (int nt = 0; nt < 4; ++nt) {
    int sgn = n0 + nt * 16 + l15;
    if (sgn < Ns) {
      int nb = sgn / S;
      int sr = sgn - nb * S;
      #pragma unroll
      for (int r = 0; r < 8; ++r) {
        int m = m0 + wave * 16 + r + half * 8;
        if (m < Cout)
          Y[(size_t)nb * Cout * S + (size_t)m * S + sr] = acc[nt][r];
      }
    }
  }
}

// ---------------------------------------------------------------------------
// Transposed conv, k=2, lhs_dilation=2, pad(1,1) with pre-flipped weights ==
// single-tap gather: out[z,y,x] uses in[z>>1,y>>1,x>>1] and tap (z&1,y&1,x&1)
// of the *unflipped* weight. Tiny FLOPs (~1 GFLOP) -> direct VALU kernel.
// ---------------------------------------------------------------------------
__global__ __launch_bounds__(256) void tconv3d_kernel(
    const h16* __restrict__ X, const h16* __restrict__ Wt, float* __restrict__ Y,
    int Nb, int Cin, int Cout, int Do, int Ho, int Wo)
{
  const int S = Do * Ho * Wo;
  long long total = (long long)Nb * Cout * S;
  long long idx = (long long)blockIdx.x * blockDim.x + threadIdx.x;
  if (idx >= total) return;
  int sr = (int)(idx % S);
  int t  = (int)(idx / S);
  int co = t % Cout;
  int nb = t / Cout;
  int z = sr / (Ho * Wo); int r2 = sr - z * (Ho * Wo);
  int y = r2 / Wo;        int x  = r2 - y * Wo;
  const int Di = Do >> 1, Hi = Ho >> 1, Wi = Wo >> 1;
  const int Sin = Di * Hi * Wi;
  const int tap = ((z & 1) << 2) | ((y & 1) << 1) | (x & 1);
  const h16* xin = X + (size_t)nb * Cin * Sin + ((z >> 1) * Hi + (y >> 1)) * Wi + (x >> 1);
  const h16* wp  = Wt + (size_t)co * Cin * 8 + tap;
  float acc = 0.f;
  for (int ci = 0; ci < Cin; ++ci)
    acc += (float)xin[(size_t)ci * Sin] * (float)wp[(size_t)ci * 8];
  Y[idx] = acc;
}

// ---------------------------------------------------------------------------
// BatchNorm statistics: one block per channel, deterministic LDS tree
// reduction (no float atomics -> bitwise-stable across graph replays).
// ---------------------------------------------------------------------------
__global__ __launch_bounds__(256) void bn_stats_kernel(
    const float* __restrict__ Y, float* __restrict__ stats, int Nb, int C, int S)
{
  __shared__ float s1[256];
  __shared__ float s2[256];
  const int c = blockIdx.x;
  float sum = 0.f, ss = 0.f;
  for (int n = 0; n < Nb; ++n) {
    const float* p = Y + ((size_t)n * C + c) * S;
    for (int i = threadIdx.x; i < S; i += 256) {
      float v = p[i];
      sum += v; ss += v * v;
    }
  }
  s1[threadIdx.x] = sum; s2[threadIdx.x] = ss;
  __syncthreads();
  for (int o = 128; o > 0; o >>= 1) {
    if ((int)threadIdx.x < o) {
      s1[threadIdx.x] += s1[threadIdx.x + o];
      s2[threadIdx.x] += s2[threadIdx.x + o];
    }
    __syncthreads();
  }
  if (threadIdx.x == 0) {
    float inv = 1.f / ((float)Nb * (float)S);
    float m = s1[0] * inv;
    float v = s2[0] * inv - m * m;
    stats[2 * c] = m;
    stats[2 * c + 1] = v;
  }
}

// BN apply + ELU + optional skip-add; dual write f32 (skip store) + f16 (next conv).
__global__ __launch_bounds__(256) void bn_elu_kernel(
    const float* __restrict__ X, const float* __restrict__ g, const float* __restrict__ b,
    const float* __restrict__ stats, const float* __restrict__ skip,
    float* __restrict__ outF, h16* __restrict__ outH, int Nb, int C, int S)
{
  long long total = (long long)Nb * C * S;
  long long idx = (long long)blockIdx.x * blockDim.x + threadIdx.x;
  if (idx >= total) return;
  int c = (int)((idx / S) % C);
  float m = stats[2 * c], v = stats[2 * c + 1];
  float xh = (X[idx] - m) * rsqrtf(v + 1e-5f);
  float y = g[c] * xh + b[c];
  y = y > 0.f ? y : (__expf(y) - 1.f);          // ELU, alpha = 1
  if (skip) y += skip[idx];
  if (outF) outF[idx] = y;
  if (outH) outH[idx] = (h16)y;
}

__global__ __launch_bounds__(256) void cvt_f32_h16_kernel(
    const float* __restrict__ src, h16* __restrict__ dst, long long n)
{
  long long i = (long long)blockIdx.x * blockDim.x + threadIdx.x;
  if (i < n) dst[i] = (h16)src[i];
}

// final concat: out (Nb,16,S): ch0 = dens (ch0 of dec4b post), ch1..15 = head2 post
__global__ __launch_bounds__(256) void concat_kernel(
    const float* __restrict__ densSrc, const float* __restrict__ feat,
    float* __restrict__ out, int Nb, int S)
{
  long long total = (long long)Nb * 16 * S;
  long long idx = (long long)blockIdx.x * blockDim.x + threadIdx.x;
  if (idx >= total) return;
  int sr = (int)(idx % S);
  int t  = (int)(idx / S);
  int c = t % 16, n = t / 16;
  float v = (c == 0) ? densSrc[((size_t)n * 16) * S + sr]
                     : feat[((size_t)n * 15 + (c - 1)) * S + sr];
  out[idx] = v;
}

// ---------------------------------------------------------------------------
// Host orchestration (stream launches only; graph-capture safe).
// ---------------------------------------------------------------------------
static inline void launch_conv(hipStream_t st, const h16* X, const h16* Wt, float* Y,
    int Nb, int Cin, int Ctot, int coff, int Din, int Hin, int Win,
    int Cout, int k, int stride, int padlo, int Do, int Ho, int Wo)
{
  int K = Cin * k * k * k;
  int aok32 = (K % 32 == 0) ? 1 : 0;
  long long Ns = (long long)Nb * Do * Ho * Wo;
  dim3 grid((unsigned)((Ns + 63) / 64), (unsigned)((Cout + 63) / 64));
  conv3d_wmma_kernel<<<grid, 128, 0, st>>>(X, Wt, Y, Nb, Cin, Ctot, coff,
      Din, Hin, Win, Cout, K, k, k * k, k * k * k, stride, padlo, Do, Ho, Wo, aok32);
}

static inline void launch_bn_elu(hipStream_t st, const float* Xc, const float* g,
    const float* b, float* stats, const float* skip, float* outF, h16* outH,
    int Nb, int C, int S)
{
  bn_stats_kernel<<<C, 256, 0, st>>>(Xc, stats, Nb, C, S);
  long long total = (long long)Nb * C * S;
  bn_elu_kernel<<<(unsigned)((total + 255) / 256), 256, 0, st>>>(
      Xc, g, b, stats, skip, outF, outH, Nb, C, S);
}

extern "C" void kernel_launch(void* const* d_in, const int* in_sizes, int n_in,
                              void* d_out, int out_size, void* d_ws, size_t ws_size,
                              hipStream_t stream)
{
  (void)in_sizes; (void)n_in; (void)out_size; (void)ws_size;
  const float* x_in = (const float*)d_in[0];
  // d_in: [0]=partial_in, [1..3]=unused feats, then (w,g,b) per layer.
  auto Wf = [&](int i) { return (const float*)d_in[4 + 3 * i]; };
  auto Gf = [&](int i) { return (const float*)d_in[4 + 3 * i + 1]; };
  auto Bf = [&](int i) { return (const float*)d_in[4 + 3 * i + 2]; };

  // layer table (o, i, k) in reference order
  static const int LO[17] = {16,64,64,128,128,512,512,128,128,64,64,16,16,16,16,15,15};
  static const int LI[17] = {16,16,64, 64,128,128,512,512,128,128,64,64,16,16,16,15,15};
  static const int LK[17] = { 3, 2, 3,  2,  3,  2,  3,  4,  3,  2, 3, 2, 3, 2, 3, 2, 3};

  char* ws = (char*)d_ws;
  size_t off = 0;
  auto alloc = [&](size_t bytes) -> char* {
    char* p = ws + off;
    off = (off + bytes + 255) & ~(size_t)255;
    return p;
  };

  const int Nb = 2;
  const int S64 = 64 * 64 * 64, S32 = 32 * 32 * 32, S16 = 16 * 16 * 16;

  float* encs1F = (float*)alloc((size_t)Nb * 16  * S64 * 4);  // skip @ stride 1
  float* encs4F = (float*)alloc((size_t)Nb * 64  * S32 * 4);  // skip @ stride 4
  float* encs8F = (float*)alloc((size_t)Nb * 128 * S16 * 4);  // skip @ stride 8
  float* tmpA   = (float*)alloc((size_t)Nb * 16  * S64 * 4);  // conv raw output
  float* tmpB   = (float*)alloc((size_t)Nb * 16  * S64 * 4);  // dec4b post (dens source)
  float* featF  = (float*)alloc((size_t)Nb * 15  * S64 * 4);  // head2 post
  h16*   hA     = (h16*)  alloc((size_t)Nb * 16  * S64 * 2);  // f16 activation ping
  h16*   hB     = (h16*)  alloc((size_t)Nb * 16  * S64 * 2);  // f16 activation pong
  float* stats  = (float*)alloc(512 * 2 * 4);

  h16* wh[17];
  for (int i = 0; i < 17; ++i) {
    size_t n = (size_t)LO[i] * LI[i] * LK[i] * LK[i] * LK[i];
    wh[i] = (h16*)alloc(n * 2);
  }

  // convert weights + input to f16
  for (int i = 0; i < 17; ++i) {
    long long n = (long long)LO[i] * LI[i] * LK[i] * LK[i] * LK[i];
    cvt_f32_h16_kernel<<<(unsigned)((n + 255) / 256), 256, 0, stream>>>(Wf(i), wh[i], n);
  }
  {
    long long n = (long long)Nb * 16 * S64;
    cvt_f32_h16_kernel<<<(unsigned)((n + 255) / 256), 256, 0, stream>>>(x_in, hA, n);
  }

  // ---- encoder ----
  // enc1: 16->16 k3 pad1 @64^3
  launch_conv(stream, hA, wh[0], tmpA, Nb, 16, 16, 0, 64,64,64, 16, 3,1,1, 64,64,64);
  launch_bn_elu(stream, tmpA, Gf(0), Bf(0), stats, nullptr, encs1F, hB, Nb, 16, S64);
  // enc2a: 16->64 k2 s2 pad0, 64^3 -> 32^3
  launch_conv(stream, hB, wh[1], tmpA, Nb, 16, 16, 0, 64,64,64, 64, 2,2,0, 32,32,32);
  launch_bn_elu(stream, tmpA, Gf(1), Bf(1), stats, nullptr, nullptr, hA, Nb, 64, S32);
  // enc2b: 64->64 k3 pad1 @32^3
  launch_conv(stream, hA, wh[2], tmpA, Nb, 64, 64, 0, 32,32,32, 64, 3,1,1, 32,32,32);
  launch_bn_elu(stream, tmpA, Gf(2), Bf(2), stats, nullptr, encs4F, hB, Nb, 64, S32);
  // enc3a: 64->128 k2 s2 pad0, 32^3 -> 16^3
  launch_conv(stream, hB, wh[3], tmpA, Nb, 64, 64, 0, 32,32,32, 128, 2,2,0, 16,16,16);
  launch_bn_elu(stream, tmpA, Gf(3), Bf(3), stats, nullptr, nullptr, hA, Nb, 128, S16);
  // enc3b: 128->128 k3 pad1 @16^3
  launch_conv(stream, hA, wh[4], tmpA, Nb, 128, 128, 0, 16,16,16, 128, 3,1,1, 16,16,16);
  launch_bn_elu(stream, tmpA, Gf(4), Bf(4), stats, nullptr, encs8F, hB, Nb, 128, S16);
  // enc4a: 128->512 k2 s1 pad(0,1) @16^3
  launch_conv(stream, hB, wh[5], tmpA, Nb, 128, 128, 0, 16,16,16, 512, 2,1,0, 16,16,16);
  launch_bn_elu(stream, tmpA, Gf(5), Bf(5), stats, nullptr, nullptr, hA, Nb, 512, S16);
  // enc4b: 512->512 k3 pad1 @16^3  (dominant layer, ~116 GFLOP)
  launch_conv(stream, hA, wh[6], tmpA, Nb, 512, 512, 0, 16,16,16, 512, 3,1,1, 16,16,16);
  launch_bn_elu(stream, tmpA, Gf(6), Bf(6), stats, nullptr, nullptr, hB, Nb, 512, S16);

  // ---- decoder ----
  // dec1a: 512->128 k4 pad(1,2) @16^3
  launch_conv(stream, hB, wh[7], tmpA, Nb, 512, 512, 0, 16,16,16, 128, 4,1,1, 16,16,16);
  launch_bn_elu(stream, tmpA, Gf(7), Bf(7), stats, nullptr, nullptr, hA, Nb, 128, S16);
  // dec1b: 128->128 k3 pad1 @16^3, + enc_s8 skip
  launch_conv(stream, hA, wh[8], tmpA, Nb, 128, 128, 0, 16,16,16, 128, 3,1,1, 16,16,16);
  launch_bn_elu(stream, tmpA, Gf(8), Bf(8), stats, encs8F, nullptr, hB, Nb, 128, S16);
  // dec2a: tconv 128->64, 16^3 -> 32^3
  {
    long long total = (long long)Nb * 64 * S32;
    tconv3d_kernel<<<(unsigned)((total + 255) / 256), 256, 0, stream>>>(
        hB, wh[9], tmpA, Nb, 128, 64, 32, 32, 32);
  }
  launch_bn_elu(stream, tmpA, Gf(9), Bf(9), stats, nullptr, nullptr, hA, Nb, 64, S32);
  // dec2b: 64->64 k3 pad1 @32^3, + enc_s4 skip
  launch_conv(stream, hA, wh[10], tmpA, Nb, 64, 64, 0, 32,32,32, 64, 3,1,1, 32,32,32);
  launch_bn_elu(stream, tmpA, Gf(10), Bf(10), stats, encs4F, nullptr, hB, Nb, 64, S32);
  // dec3a: tconv 64->16, 32^3 -> 64^3
  {
    long long total = (long long)Nb * 16 * S64;
    tconv3d_kernel<<<(unsigned)((total + 255) / 256), 256, 0, stream>>>(
        hB, wh[11], tmpA, Nb, 64, 16, 64, 64, 64);
  }
  launch_bn_elu(stream, tmpA, Gf(11), Bf(11), stats, nullptr, nullptr, hA, Nb, 16, S64);
  // dec3b: 16->16 k3 pad1 @64^3, + enc_s1 skip
  launch_conv(stream, hA, wh[12], tmpA, Nb, 16, 16, 0, 64,64,64, 16, 3,1,1, 64,64,64);
  launch_bn_elu(stream, tmpA, Gf(12), Bf(12), stats, encs1F, nullptr, hB, Nb, 16, S64);
  // dec4a: 16->16 k2 pad(0,1) @64^3
  launch_conv(stream, hB, wh[13], tmpA, Nb, 16, 16, 0, 64,64,64, 16, 2,1,0, 64,64,64);
  launch_bn_elu(stream, tmpA, Gf(13), Bf(13), stats, nullptr, nullptr, hA, Nb, 16, S64);
  // dec4b: 16->16 k3 pad1 @64^3 -> tmpB (f32, dens source) + hB (f16, head input)
  launch_conv(stream, hA, wh[14], tmpA, Nb, 16, 16, 0, 64,64,64, 16, 3,1,1, 64,64,64);
  launch_bn_elu(stream, tmpA, Gf(14), Bf(14), stats, nullptr, tmpB, hB, Nb, 16, S64);

  // ---- head on channels 1..15 ----
  // head1: Cin=15 (coff=1, Ctot=16) -> 15, k2 pad(0,1) @64^3
  launch_conv(stream, hB, wh[15], tmpA, Nb, 15, 16, 1, 64,64,64, 15, 2,1,0, 64,64,64);
  launch_bn_elu(stream, tmpA, Gf(15), Bf(15), stats, nullptr, nullptr, hA, Nb, 15, S64);
  // head2: 15->15 k3 pad1 @64^3 -> featF (f32)
  launch_conv(stream, hA, wh[16], tmpA, Nb, 15, 15, 0, 64,64,64, 15, 3,1,1, 64,64,64);
  launch_bn_elu(stream, tmpA, Gf(16), Bf(16), stats, nullptr, featF, nullptr, Nb, 15, S64);

  // ---- concat [dens, feat] into d_out (Nb,16,64^3) f32 ----
  {
    long long total = (long long)Nb * 16 * S64;
    concat_kernel<<<(unsigned)((total + 255) / 256), 256, 0, stream>>>(
        tmpB, featF, (float*)d_out, Nb, S64);
  }
}
